// LiquidFusion_10651518894285
// MI455X (gfx1250) — matrix operand order repaired and checked by reading, not compiled, not run on hardware
//
#include <hip/hip_runtime.h>
#include <hip/hip_bf16.h>

typedef __attribute__((ext_vector_type(16))) __bf16 v16bf;
typedef __attribute__((ext_vector_type(8)))  __bf16 v8bf;
typedef __attribute__((ext_vector_type(8)))  float  v8f;

#define HIDDEN   2048
#define NHEADS   16
#define HEADDIM  128
#define SEQ      2048
#define BATCH    2
#define BH       (BATCH * NHEADS)      // 32
#define ROWS     (BATCH * SEQ)         // 4096
#define SINKSZ   4
#define HEAVY_BUDGET  409              // int(0.2 * 2048)
#define RECENT_BUDGET 614              // int(0.3 * 2048)
#define INV_SQRT_D 0.08838834764831845f
#define NEGINF (-3.0e38f)

static __device__ __forceinline__ v8f zero8() {
  v8f z;
#pragma unroll
  for (int i = 0; i < 8; ++i) z[i] = 0.0f;
  return z;
}

static __device__ __forceinline__ v16bf cat16(v8bf lo, v8bf hi) {
  return __builtin_shufflevector(lo, hi, 0,1,2,3,4,5,6,7,8,9,10,11,12,13,14,15);
}

static __device__ __forceinline__ v8f bfwmma(v16bf a, v16bf b, v8f c) {
  return __builtin_amdgcn_wmma_f32_16x16x32_bf16(false, a, false, b, (short)0, c,
                                                 false, false);
}

// CDNA5 async global->LDS copy (no VGPR staging, tracked by ASYNCcnt).
// the LDS destination address comes from the vdst VGPR; flat->LDS mapping uses addr[31:0], so the
// low 32 bits of a generic pointer to a __shared__ object are the LDS offset.
static __device__ __forceinline__ void async_b128(unsigned lds_off, const void* g) {
  asm volatile("global_load_async_to_lds_b128 %0, %1, off"
               :: "v"(lds_off), "v"(g) : "memory");
}
static __device__ __forceinline__ void wait_async_le4() {
  asm volatile("s_wait_asynccnt 0x4" ::: "memory");
}
static __device__ __forceinline__ void wait_async_0() {
  asm volatile("s_wait_asynccnt 0x0" ::: "memory");
}

// reductions across the 16 lanes of each half-wave (B/C frag column groups)
static __device__ __forceinline__ float half16_max(float x) {
  x = fmaxf(x, __shfl_xor(x, 1, 32));
  x = fmaxf(x, __shfl_xor(x, 2, 32));
  x = fmaxf(x, __shfl_xor(x, 4, 32));
  x = fmaxf(x, __shfl_xor(x, 8, 32));
  return x;
}
static __device__ __forceinline__ float half16_sum(float x) {
  x += __shfl_xor(x, 1, 32);
  x += __shfl_xor(x, 2, 32);
  x += __shfl_xor(x, 4, 32);
  x += __shfl_xor(x, 8, 32);
  return x;
}

// ---------------------------------------------------------------- convert
__global__ void k_f32_to_bf16(const float* __restrict__ in,
                              __bf16* __restrict__ out, int n) {
  int i = blockIdx.x * blockDim.x + threadIdx.x;
  int stride = gridDim.x * blockDim.x;
  for (; i < n; i += stride) out[i] = (__bf16)in[i];
}

// ---------------------------------------------------------------- GEMM C = A * W^T
// A: [M][K] bf16 row-major.  W: [N][K] bf16 row-major.  C: [M][N] fp32.
// 128x128 block tile, 8 waves (4Mx2N), 32x64 per wave (8 WMMAs per K-step).
// Double-buffered LDS staging via global_load_async_to_lds_b128:
// issue tile t+1, s_wait_asynccnt 4 (in-order completion => tile t resident),
// barrier, 8 WMMAs from LDS, barrier.
#define GT_M 128
#define GT_N 128
#define GT_K 32
#define LPAD 8
#define SROW (GT_K + LPAD)
__global__ __launch_bounds__(256) void k_gemm_xwt(
    const __bf16* __restrict__ A, const __bf16* __restrict__ W,
    float* __restrict__ C, int M, int N, int K) {
  __shared__ __bf16 sA[2][GT_M][SROW];
  __shared__ __bf16 sW[2][GT_N][SROW];
  const int tid  = threadIdx.x;
  const int lane = tid & 31;
  const int wave = tid >> 5;          // 0..7
  const int wm   = wave >> 1;         // 0..3  -> M offset wm*32
  const int wn   = wave & 1;          // 0..1  -> N offset wn*64
  const int half = lane >> 4;
  const int r    = lane & 15;
  const int m0 = blockIdx.y * GT_M;
  const int n0 = blockIdx.x * GT_N;

  const int arow = tid >> 1, acol = (tid & 1) * 16;   // A/W tiles: 128 rows x 32

  const __bf16* gA = &A[(size_t)(m0 + arow) * K + acol];
  const __bf16* gW = &W[(size_t)(n0 + arow) * K + acol];

  v8f acc[2][4];
#pragma unroll
  for (int i = 0; i < 2; ++i)
#pragma unroll
    for (int j = 0; j < 4; ++j) acc[i][j] = zero8();

  auto issue = [&](int buf, int k0) {
    async_b128((unsigned)(uintptr_t)&sA[buf][arow][acol],     gA + k0);
    async_b128((unsigned)(uintptr_t)&sA[buf][arow][acol + 8], gA + k0 + 8);
    async_b128((unsigned)(uintptr_t)&sW[buf][arow][acol],     gW + k0);
    async_b128((unsigned)(uintptr_t)&sW[buf][arow][acol + 8], gW + k0 + 8);
  };

  issue(0, 0);
  const int T = K / GT_K;
  for (int t = 0; t < T; ++t) {
    const int cur = t & 1;
    if (t + 1 < T) {
      issue(cur ^ 1, (t + 1) * GT_K);
      wait_async_le4();     // tile t's 4 copies complete (in-order)
    } else {
      wait_async_0();
    }
    __syncthreads();
    v16bf bf[4];
#pragma unroll
    for (int bi = 0; bi < 4; ++bi) {
      const int nrow = wn * 64 + bi * 16 + r;
      bf[bi] = cat16(*(const v8bf*)&sW[cur][nrow][half * 8],
                     *(const v8bf*)&sW[cur][nrow][16 + half * 8]);
    }
#pragma unroll
    for (int ai = 0; ai < 2; ++ai) {
      const int mrow = wm * 32 + ai * 16 + r;
      v16bf a = cat16(*(const v8bf*)&sA[cur][mrow][half * 8],
                      *(const v8bf*)&sA[cur][mrow][16 + half * 8]);
#pragma unroll
      for (int bi = 0; bi < 4; ++bi)
        acc[ai][bi] = bfwmma(a, bf[bi], acc[ai][bi]);
    }
    __syncthreads();
  }
#pragma unroll
  for (int ai = 0; ai < 2; ++ai)
#pragma unroll
    for (int bi = 0; bi < 4; ++bi)
#pragma unroll
      for (int v = 0; v < 8; ++v) {
        const int row = m0 + wm * 32 + ai * 16 + v + 8 * half;
        const int col = n0 + wn * 64 + bi * 16 + r;
        C[(size_t)row * N + col] = acc[ai][bi][v];
      }
}

// ---------------------------------------------------------------- RoPE + layout pack
// Qf/Kf/Vf: [B*S][HIDDEN] fp32.  qbf/kbf: [bh][s][d] bf16.  vt: [bh][d][s] bf16.
__global__ void k_rope_pack(const float* __restrict__ Qf, const float* __restrict__ Kf,
                            const float* __restrict__ Vf, const int* __restrict__ pos,
                            __bf16* __restrict__ qbf, __bf16* __restrict__ kbf,
                            __bf16* __restrict__ vt) {
  const int idx = blockIdx.x * blockDim.x + threadIdx.x;
  if (idx >= ROWS * HIDDEN) return;
  const int d   = idx & (HEADDIM - 1);
  const int h   = (idx >> 7) & (NHEADS - 1);
  const int row = idx >> 11;              // b*SEQ + s
  const int s   = row & (SEQ - 1);
  const int b   = row >> 11;
  const int bh  = b * NHEADS + h;

  const int fi = d & 63;
  const float inv = __powf(10000.0f, -(float)(2 * fi) / (float)HEADDIM);
  const float ang = (float)pos[row] * inv;
  const float c = __cosf(ang), sn = __sinf(ang);
  const int src2 = idx + ((d < 64) ? 64 : -64);

  const float q  = Qf[idx], qp = Qf[src2];
  const float k  = Kf[idx], kp = Kf[src2];
  const float qr = q * c + ((d < 64) ? -qp : qp) * sn;
  const float kr = k * c + ((d < 64) ? -kp : kp) * sn;

  qbf[((size_t)bh * SEQ + s) * HEADDIM + d] = (__bf16)qr;
  kbf[((size_t)bh * SEQ + s) * HEADDIM + d] = (__bf16)kr;
  vt [((size_t)bh * HEADDIM + d) * SEQ + s] = (__bf16)Vf[idx];
}

// ---------------------------------------------------------------- pass 1: row max / sumexp of full QK^T
__global__ __launch_bounds__(256) void k_rowstats(
    const __bf16* __restrict__ qbf, const __bf16* __restrict__ kbf,
    float* __restrict__ rowm, float* __restrict__ rowl) {
  const int bh   = blockIdx.y;
  const int wave = threadIdx.x >> 5, lane = threadIdx.x & 31;
  const int half = lane >> 4, r = lane & 15;
  const int q0   = blockIdx.x * 128 + wave * 16;
  const __bf16* Q  = qbf + (size_t)bh * SEQ * HEADDIM;
  const __bf16* Kp = kbf + (size_t)bh * SEQ * HEADDIM;

  v16bf qa[4];
#pragma unroll
  for (int c = 0; c < 4; ++c) {
    const __bf16* p = Q + (size_t)(q0 + r) * HEADDIM + c * 32 + half * 8;
    qa[c] = cat16(*(const v8bf*)p, *(const v8bf*)(p + 16));
  }
  float m[8], l[8];
#pragma unroll
  for (int v = 0; v < 8; ++v) { m[v] = NEGINF; l[v] = 0.0f; }

  for (int kb = 0; kb < SEQ; kb += 16) {
    v8f sc = zero8();
#pragma unroll
    for (int c = 0; c < 4; ++c) {
      const __bf16* kr = Kp + (size_t)(kb + r) * HEADDIM + c * 32 + half * 8;
      sc = bfwmma(qa[c], cat16(*(const v8bf*)kr, *(const v8bf*)(kr + 16)), sc);
    }
#pragma unroll
    for (int v = 0; v < 8; ++v) {
      const float s  = sc[v] * INV_SQRT_D;
      const float nm = fmaxf(m[v], half16_max(s));
      const float rs = half16_sum(__expf(s - nm));
      l[v] = l[v] * __expf(m[v] - nm) + rs;
      m[v] = nm;
    }
  }
  if (r == 0) {
#pragma unroll
    for (int v = 0; v < 8; ++v) {
      const int row = q0 + v + 8 * half;
      rowm[(size_t)bh * SEQ + row] = m[v];
      rowl[(size_t)bh * SEQ + row] = l[v];
    }
  }
}

// ---------------------------------------------------------------- pass 2: column sums of tmp softmax
__global__ __launch_bounds__(256) void k_colsum(
    const __bf16* __restrict__ qbf, const __bf16* __restrict__ kbf,
    const float* __restrict__ rowm, const float* __restrict__ rowl,
    float* __restrict__ colsum) {
  const int bh   = blockIdx.y;
  const int wave = threadIdx.x >> 5, lane = threadIdx.x & 31;
  const int half = lane >> 4, r = lane & 15;
  const int q0   = blockIdx.x * 128 + wave * 16;
  const __bf16* Q  = qbf + (size_t)bh * SEQ * HEADDIM;
  const __bf16* Kp = kbf + (size_t)bh * SEQ * HEADDIM;

  v16bf qa[4];
#pragma unroll
  for (int c = 0; c < 4; ++c) {
    const __bf16* p = Q + (size_t)(q0 + r) * HEADDIM + c * 32 + half * 8;
    qa[c] = cat16(*(const v8bf*)p, *(const v8bf*)(p + 16));
  }
  float m[8], linv[8];
#pragma unroll
  for (int v = 0; v < 8; ++v) {
    const int row = q0 + v + 8 * half;
    m[v] = rowm[(size_t)bh * SEQ + row];
    const float L = rowl[(size_t)bh * SEQ + row];
    linv[v] = (L > 0.0f) ? 1.0f / L : 0.0f;
  }
  for (int kb = 0; kb < SEQ; kb += 16) {
    v8f sc = zero8();
#pragma unroll
    for (int c = 0; c < 4; ++c) {
      const __bf16* kr = Kp + (size_t)(kb + r) * HEADDIM + c * 32 + half * 8;
      sc = bfwmma(qa[c], cat16(*(const v8bf*)kr, *(const v8bf*)(kr + 16)), sc);
    }
    float cs = 0.0f;
#pragma unroll
    for (int v = 0; v < 8; ++v)
      cs += __expf(sc[v] * INV_SQRT_D - m[v]) * linv[v];
    cs += __shfl_xor(cs, 16, 32);
    if (half == 0) atomicAdd(&colsum[(size_t)bh * SEQ + kb + r], cs);
  }
}

// ---------------------------------------------------------------- top-k heavy mask
__global__ __launch_bounds__(256) void k_topk(const float* __restrict__ colsum,
                                              unsigned* __restrict__ hmask) {
  __shared__ float    vals[SEQ];
  __shared__ unsigned mask[SEQ / 32];
  __shared__ float    rv[256];
  __shared__ int      ri[256];
  const int bh = blockIdx.x, t = threadIdx.x;
  for (int i = t; i < SEQ; i += 256) vals[i] = colsum[(size_t)bh * SEQ + i];
  for (int i = t; i < SEQ / 32; i += 256) mask[i] = 0u;
  __syncthreads();
  for (int it = 0; it < HEAVY_BUDGET; ++it) {
    float bv = NEGINF; int bi = SEQ;
    for (int i = t; i < SEQ; i += 256) {
      const float x = vals[i];
      if (x > bv || (x == bv && i < bi)) { bv = x; bi = i; }
    }
    rv[t] = bv; ri[t] = bi;
    __syncthreads();
    for (int off = 128; off > 0; off >>= 1) {
      if (t < off) {
        const float xv = rv[t + off]; const int xi = ri[t + off];
        if (xv > rv[t] || (xv == rv[t] && xi < ri[t])) { rv[t] = xv; ri[t] = xi; }
      }
      __syncthreads();
    }
    if (t == 0) { const int w = ri[0]; mask[w >> 5] |= (1u << (w & 31)); vals[w] = NEGINF; }
    __syncthreads();
  }
  for (int i = t; i < SEQ / 32; i += 256) hmask[(size_t)bh * (SEQ / 32) + i] = mask[i];
}

// ---------------------------------------------------------------- final masked attention (flash-style)
__global__ __launch_bounds__(256) void k_attn(
    const __bf16* __restrict__ qbf, const __bf16* __restrict__ kbf,
    const __bf16* __restrict__ vt, const unsigned* __restrict__ hmask,
    __bf16* __restrict__ attn) {
  __shared__ unsigned smask[SEQ / 32];
  __shared__ __bf16 sP[8][16][32 + LPAD];
  const int bh = blockIdx.y;
  const int b = bh / NHEADS, h = bh % NHEADS;
  const int wave = threadIdx.x >> 5, lane = threadIdx.x & 31;
  const int half = lane >> 4, r = lane & 15;
  const int q0 = blockIdx.x * 128 + wave * 16;

  for (int i = threadIdx.x; i < SEQ / 32; i += 256)
    smask[i] = hmask[(size_t)bh * (SEQ / 32) + i];
  __syncthreads();

  const __bf16* Q  = qbf + (size_t)bh * SEQ * HEADDIM;
  const __bf16* Kp = kbf + (size_t)bh * SEQ * HEADDIM;
  const __bf16* V  = vt  + (size_t)bh * HEADDIM * SEQ;

  v16bf qa[4];
#pragma unroll
  for (int c = 0; c < 4; ++c) {
    const __bf16* p = Q + (size_t)(q0 + r) * HEADDIM + c * 32 + half * 8;
    qa[c] = cat16(*(const v8bf*)p, *(const v8bf*)(p + 16));
  }
  float m[8], l[8]; v8f oacc[8];
#pragma unroll
  for (int v = 0; v < 8; ++v) { m[v] = NEGINF; l[v] = 0.0f; }
#pragma unroll
  for (int nf = 0; nf < 8; ++nf) oacc[nf] = zero8();

  // step 0 = sink keys (k[0:4], unmasked); steps 1..64 = regular 32-key tiles
  for (int step = 0; step <= SEQ / 32; ++step) {
    const bool sink = (step == 0);
    const int  kb   = sink ? 0 : (step - 1) * 32;
    v8f sc[2];
#pragma unroll
    for (int kt = 0; kt < 2; ++kt) {
      sc[kt] = zero8();
#pragma unroll
      for (int c = 0; c < 4; ++c) {
        const __bf16* kr = Kp + (size_t)(kb + kt * 16 + r) * HEADDIM + c * 32 + half * 8;
        sc[kt] = bfwmma(qa[c], cat16(*(const v8bf*)kr, *(const v8bf*)(kr + 16)), sc[kt]);
      }
    }
#pragma unroll
    for (int kt = 0; kt < 2; ++kt) {
      const int key = kb + kt * 16 + r;
#pragma unroll
      for (int v = 0; v < 8; ++v) {
        const int qrow = q0 + v + 8 * half;
        const bool valid = sink ? (kt == 0 && r < SINKSZ)
                                : (((smask[key >> 5] >> (key & 31)) & 1u) ||
                                   (key - qrow <= RECENT_BUDGET));
        sc[kt][v] = valid ? sc[kt][v] * INV_SQRT_D : NEGINF;
      }
    }
#pragma unroll
    for (int v = 0; v < 8; ++v) {
      const float nm = fmaxf(m[v], half16_max(fmaxf(sc[0][v], sc[1][v])));
      const float sf = __expf(m[v] - nm);
      const float p0 = __expf(sc[0][v] - nm);
      const float p1 = __expf(sc[1][v] - nm);
      l[v] = l[v] * sf + half16_sum(p0 + p1);
      m[v] = nm;
#pragma unroll
      for (int nf = 0; nf < 8; ++nf) oacc[nf][v] *= sf;
      const int row = v + 8 * half;
      sP[wave][row][r]      = (__bf16)p0;
      sP[wave][row][16 + r] = (__bf16)p1;
    }
    __syncthreads();
    // O += P @ V   (P: 16x32 A-frag from LDS; V^T rows give contiguous B-frags)
    v16bf pa = cat16(*(const v8bf*)&sP[wave][r][half * 8],
                     *(const v8bf*)&sP[wave][r][16 + half * 8]);
#pragma unroll
    for (int nf = 0; nf < 8; ++nf) {
      const __bf16* vr = V + (size_t)(nf * 16 + r) * SEQ + kb + half * 8;
      oacc[nf] = bfwmma(pa, cat16(*(const v8bf*)vr, *(const v8bf*)(vr + 16)), oacc[nf]);
    }
    __syncthreads();
  }
#pragma unroll
  for (int v = 0; v < 8; ++v) {
    const int qrow = q0 + v + 8 * half;
    const float inv = (l[v] > 0.0f) ? 1.0f / l[v] : 0.0f;
    const size_t base = ((size_t)b * SEQ + qrow) * HIDDEN + (size_t)h * HEADDIM;
#pragma unroll
    for (int nf = 0; nf < 8; ++nf)
      attn[base + nf * 16 + r] = (__bf16)(oacc[nf][v] * inv);
  }
}

// ---------------------------------------------------------------- host
extern "C" void kernel_launch(void* const* d_in, const int* in_sizes, int n_in,
                              void* d_out, int out_size, void* d_ws, size_t ws_size,
                              hipStream_t stream) {
  const float* X   = (const float*)d_in[0];
  const float* Wq  = (const float*)d_in[1];
  const float* Wk  = (const float*)d_in[2];
  const float* Wv  = (const float*)d_in[3];
  const float* Wo  = (const float*)d_in[4];
  const int*   pos = (const int*)  d_in[5];
  float* out = (float*)d_out;

  char* ws = (char*)d_ws;
  size_t off = 0;
  auto take = [&](size_t bytes) -> char* {
    char* p = ws + off;
    off += (bytes + 255) & ~(size_t)255;
    return p;
  };
  __bf16* Xbf  = (__bf16*)take((size_t)ROWS * HIDDEN * 2);
  __bf16* Wqb  = (__bf16*)take((size_t)HIDDEN * HIDDEN * 2);
  __bf16* Wkb  = (__bf16*)take((size_t)HIDDEN * HIDDEN * 2);
  __bf16* Wvb  = (__bf16*)take((size_t)HIDDEN * HIDDEN * 2);
  __bf16* Wob  = (__bf16*)take((size_t)HIDDEN * HIDDEN * 2);
  float*  Qf   = (float*) take((size_t)ROWS * HIDDEN * 4);
  float*  Kf   = (float*) take((size_t)ROWS * HIDDEN * 4);
  float*  Vf   = (float*) take((size_t)ROWS * HIDDEN * 4);
  __bf16* qb   = (__bf16*)take((size_t)BH * SEQ * HEADDIM * 2);
  __bf16* kb   = (__bf16*)take((size_t)BH * SEQ * HEADDIM * 2);
  __bf16* vtb  = (__bf16*)take((size_t)BH * HEADDIM * SEQ * 2);
  float*  rowm = (float*) take((size_t)BH * SEQ * 4);
  float*  rowl = (float*) take((size_t)BH * SEQ * 4);
  float*  csum = (float*) take((size_t)BH * SEQ * 4);
  unsigned* hm = (unsigned*)take((size_t)BH * (SEQ / 32) * 4);
  __bf16* attb = (__bf16*)take((size_t)ROWS * HIDDEN * 2);
  (void)ws_size; (void)n_in; (void)in_sizes; (void)out_size;

  k_f32_to_bf16<<<2048, 256, 0, stream>>>(X,  Xbf, ROWS * HIDDEN);
  k_f32_to_bf16<<<1024, 256, 0, stream>>>(Wq, Wqb, HIDDEN * HIDDEN);
  k_f32_to_bf16<<<1024, 256, 0, stream>>>(Wk, Wkb, HIDDEN * HIDDEN);
  k_f32_to_bf16<<<1024, 256, 0, stream>>>(Wv, Wvb, HIDDEN * HIDDEN);
  k_f32_to_bf16<<<1024, 256, 0, stream>>>(Wo, Wob, HIDDEN * HIDDEN);

  dim3 gg(HIDDEN / GT_N, ROWS / GT_M);
  k_gemm_xwt<<<gg, 256, 0, stream>>>(Xbf, Wqb, Qf, ROWS, HIDDEN, HIDDEN);
  k_gemm_xwt<<<gg, 256, 0, stream>>>(Xbf, Wkb, Kf, ROWS, HIDDEN, HIDDEN);
  k_gemm_xwt<<<gg, 256, 0, stream>>>(Xbf, Wvb, Vf, ROWS, HIDDEN, HIDDEN);

  k_rope_pack<<<(ROWS * HIDDEN) / 256, 256, 0, stream>>>(Qf, Kf, Vf, pos, qb, kb, vtb);

  hipMemsetAsync(csum, 0, (size_t)BH * SEQ * 4, stream);
  dim3 ga(SEQ / 128, BH);
  k_rowstats<<<ga, 256, 0, stream>>>(qb, kb, rowm, rowl);
  k_colsum  <<<ga, 256, 0, stream>>>(qb, kb, rowm, rowl, csum);
  k_topk    <<<BH, 256, 0, stream>>>(csum, hm);
  k_attn    <<<ga, 256, 0, stream>>>(qb, kb, vtb, hm, attb);

  k_gemm_xwt<<<gg, 256, 0, stream>>>(attb, Wob, out, ROWS, HIDDEN, HIDDEN);
}
